// TinyTransformer_2345052144292
// MI455X (gfx1250) — matrix operand order repaired and checked by reading, not compiled, not run on hardware
//
#include <hip/hip_runtime.h>
#include <hip/hip_bf16.h>
#include <math.h>

// ---------------------------------------------------------------------------
// TinyTransformer forward for MI455X (gfx1250, wave32, WMMA).
// All matmuls: v_wmma_f32_16x16x32_bf16, fp32 accumulate.
// Operands pre-converted to bf16, laid out so every WMMA fragment is a
// contiguous 128-bit load (A: [M,K] row-major, B: [N,K] "transposed").
// K is a template parameter so all strides fold into immediate offsets.
// ---------------------------------------------------------------------------

#define BB 2
#define SS 2048
#define DD 512
#define HH 16
#define HDD 32
#define LL 2
#define VV 32000
#define DFF 2048
#define BS (BB * SS)     // 4096 rows
#define EPSLN 1e-5f

typedef __attribute__((ext_vector_type(16))) __bf16 v16bf;
typedef __attribute__((ext_vector_type(8)))  __bf16 v8bf;
typedef __attribute__((ext_vector_type(8)))  float  v8f;
typedef __attribute__((ext_vector_type(4)))  unsigned int v4u;
typedef __attribute__((ext_vector_type(4)))  int v4i;
typedef __attribute__((ext_vector_type(8)))  int v8i;

__device__ __forceinline__ __bf16 f2bf(float f) {
    unsigned u = __builtin_bit_cast(unsigned, f);
    u += 0x7fffu + ((u >> 16) & 1u);          // round-to-nearest-even
    unsigned short h = (unsigned short)(u >> 16);
    return __builtin_bit_cast(__bf16, h);
}

__device__ __forceinline__ v8f zero8() {
    v8f z;
#pragma unroll
    for (int i = 0; i < 8; ++i) z[i] = 0.0f;
    return z;
}

__device__ __forceinline__ v16bf cat8(v8bf lo, v8bf hi) {
    return __builtin_shufflevector(lo, hi, 0, 1, 2, 3, 4, 5, 6, 7,
                                           8, 9, 10, 11, 12, 13, 14, 15);
}

// ===========================================================================
// Embedding gather
// ===========================================================================
__global__ void embed_kernel(const int* __restrict__ tokens,
                             const float* __restrict__ emb,
                             float* __restrict__ x) {
    int row = blockIdx.x;
    int tok = tokens[row];
    const float* src = emb + (size_t)tok * DD;
    float* dst = x + (size_t)row * DD;
    for (int i = threadIdx.x; i < DD; i += blockDim.x) dst[i] = src[i];
}

// ===========================================================================
// Weight transpose+convert: W[K,N] f32 -> Wt[N,K] bf16 (32x32 LDS tiles)
// ===========================================================================
__global__ void wt_kernel(const float* __restrict__ W,
                          __bf16* __restrict__ Wt, int K, int N) {
    __shared__ float tile[32][33];
    int n0 = blockIdx.x << 5, k0 = blockIdx.y << 5;
    int tx = threadIdx.x & 31, ty = threadIdx.x >> 5;
#pragma unroll
    for (int i = ty; i < 32; i += 8)
        tile[i][tx] = W[(size_t)(k0 + i) * N + n0 + tx];
    __syncthreads();
#pragma unroll
    for (int i = ty; i < 32; i += 8)
        Wt[(size_t)(n0 + i) * K + k0 + tx] = f2bf(tile[tx][i]);
}

// ===========================================================================
// Flat f32 -> bf16 convert
// ===========================================================================
__global__ void cvt_kernel(const float* __restrict__ in,
                           __bf16* __restrict__ out, int n) {
    int i = blockIdx.x * blockDim.x + threadIdx.x;
    if (i < n) out[i] = f2bf(in[i]);
}

// ===========================================================================
// LayerNorm: one wave per row of D=512, writes bf16.
// ===========================================================================
__global__ void ln_kernel(const float* __restrict__ x,
                          const float* __restrict__ g,
                          const float* __restrict__ b,
                          __bf16* __restrict__ out) {
    int wid  = (blockIdx.x * blockDim.x + threadIdx.x) >> 5;
    int lane = threadIdx.x & 31;
    if (wid >= BS) return;
    const float* xr = x + (size_t)wid * DD;

    float vals[16];
    float s = 0.0f;
#pragma unroll
    for (int i = 0; i < 16; ++i) { vals[i] = xr[lane + 32 * i]; s += vals[i]; }
#pragma unroll
    for (int m = 1; m < 32; m <<= 1) s += __shfl_xor(s, m, 32);
    float mu = s * (1.0f / DD);

    float vs = 0.0f;
#pragma unroll
    for (int i = 0; i < 16; ++i) { float d = vals[i] - mu; vs += d * d; }
#pragma unroll
    for (int m = 1; m < 32; m <<= 1) vs += __shfl_xor(vs, m, 32);
    float rstd = rsqrtf(vs * (1.0f / DD) + EPSLN);

    __bf16* orow = out + (size_t)wid * DD;
#pragma unroll
    for (int i = 0; i < 16; ++i) {
        int c = lane + 32 * i;
        orow[c] = f2bf((vals[i] - mu) * rstd * g[c] + b[c]);
    }
}

// ===========================================================================
// WMMA GEMM: C[M,N] = act(A[M,K]bf16 * Wt[N,K]bf16^T + bias) (+resid)
// Wave tile: 32(M) x 64(N) -> 8 wmma per 32-deep K step.
// K constexpr => all intra-tile strides are immediate offsets (no spills).
// ===========================================================================
template <int K, bool GELU, bool RESID, bool BF16OUT>
__global__ void gemm_kernel(const __bf16* __restrict__ A,
                            const __bf16* __restrict__ Wt,
                            const float* __restrict__ bias,
                            const float* __restrict__ resid,
                            void* __restrict__ Cout,
                            int M, int N) {
    int wib    = threadIdx.x >> 5;
    int lane   = threadIdx.x & 31;
    int lane16 = lane & 15;
    int hi     = lane >> 4;

    int tilesN   = N >> 6;
    int numTiles = (M >> 5) * tilesN;
    int wtile    = blockIdx.x * (blockDim.x >> 5) + wib;
    if (wtile >= numTiles) return;
    int tm = wtile / tilesN, tn = wtile % tilesN;
    int m0 = tm << 5, n0 = tn << 6;

    v8f acc[2][4];
#pragma unroll
    for (int i = 0; i < 2; ++i)
#pragma unroll
        for (int j = 0; j < 4; ++j) acc[i][j] = zero8();

    // Single base pointer each; row deltas are compile-time immediates.
    const __bf16* a0 = A  + (size_t)(m0 + lane16) * K;          // +16*K => row m0+16
    const __bf16* w0 = Wt + (size_t)(n0 + lane16) * K;          // +nt*16*K per N-tile

#pragma unroll 2
    for (int k0 = 0; k0 < K; k0 += 32) {
        // A fragments: regs 0-15 = K[hi*16..+15]
        v16bf af0 = cat8(*(const v8bf*)(a0 + k0 + (hi << 4)),
                         *(const v8bf*)(a0 + k0 + 8 + (hi << 4)));
        v16bf af1 = cat8(*(const v8bf*)(a0 + 16 * K + k0 + (hi << 4)),
                         *(const v8bf*)(a0 + 16 * K + k0 + 8 + (hi << 4)));
        __builtin_prefetch((const void*)(a0 + k0 + 32), 0, 0);
        __builtin_prefetch((const void*)(a0 + 16 * K + k0 + 32), 0, 0);
#pragma unroll
        for (int nt = 0; nt < 4; ++nt) {
            // B fragment: lane16 = n, 32 contiguous bytes of K (hi picks +16)
            v16bf bfr = *(const v16bf*)(w0 + nt * 16 * K + k0 + (hi << 4));
            __builtin_prefetch((const void*)(w0 + nt * 16 * K + k0 + 32), 0, 0);
            acc[0][nt] = __builtin_amdgcn_wmma_f32_16x16x32_bf16(
                false, af0, false, bfr, (short)0, acc[0][nt], false, false);
            acc[1][nt] = __builtin_amdgcn_wmma_f32_16x16x32_bf16(
                false, af1, false, bfr, (short)0, acc[1][nt], false, false);
        }
    }

    // epilogue: C/D layout row = mi*16 + hi*8 + v, col = n0 + nt*16 + lane16
#pragma unroll
    for (int mi = 0; mi < 2; ++mi) {
#pragma unroll
        for (int nt = 0; nt < 4; ++nt) {
            int col = n0 + (nt << 4) + lane16;
            float bv = bias[col];
#pragma unroll
            for (int v = 0; v < 8; ++v) {
                int row = m0 + (mi << 4) + (hi << 3) + v;
                float val = acc[mi][nt][v] + bv;
                if constexpr (GELU)
                    val = 0.5f * val * (1.0f + erff(val * 0.70710678118654752f));
                if constexpr (RESID)
                    val += resid[(size_t)row * N + col];
                if constexpr (BF16OUT)
                    ((__bf16*)Cout)[(size_t)row * N + col] = f2bf(val);
                else
                    ((float*)Cout)[(size_t)row * N + col] = val;
            }
        }
    }
}

// ===========================================================================
// RoPE + convert: qkv[BS,3D] f32 -> qkh[BS,2D] bf16 (q rotated | k rotated)
// ===========================================================================
__global__ void ropeconv_kernel(const float* __restrict__ qkv,
                                __bf16* __restrict__ qkh) {
    int idx = blockIdx.x * blockDim.x + threadIdx.x;
    const int TOT = BS * HH * (HDD / 2);
    if (idx >= TOT) return;
    int j   = idx & 15;
    int h   = (idx >> 4) & (HH - 1);
    int row = idx >> 8;
    int s   = row & (SS - 1);

    float theta = powf(10000.0f, -2.0f * (float)j / (float)HDD);
    float ang = (float)s * theta;
    float cs = cosf(ang), sn = sinf(ang);

    size_t sbase = (size_t)row * (3 * DD) + h * HDD + 2 * j;
    size_t dbase = (size_t)row * (2 * DD) + h * HDD + 2 * j;
    float qe = qkv[sbase], qo = qkv[sbase + 1];
    qkh[dbase]     = f2bf(cs * qe - sn * qo);
    qkh[dbase + 1] = f2bf(sn * qe + cs * qo);
    float ke = qkv[sbase + DD], ko = qkv[sbase + DD + 1];
    qkh[dbase + DD]     = f2bf(cs * ke - sn * ko);
    qkh[dbase + DD + 1] = f2bf(sn * ke + cs * ko);
}

// ===========================================================================
// V transpose+convert: qkv[:,2D:3D] f32 -> vt[b,h,d,s] bf16
// ===========================================================================
__global__ void vtrans_kernel(const float* __restrict__ qkv,
                              __bf16* __restrict__ vt) {
    int idx = blockIdx.x * blockDim.x + threadIdx.x;
    if (idx >= BS * DD) return;
    int row = idx >> 9;
    int c   = idx & (DD - 1);
    int h = c >> 5, d = c & 31;
    int b = row / SS, s = row & (SS - 1);
    vt[((size_t)((b * HH + h) * HDD + d)) * SS + s] =
        f2bf(qkv[(size_t)row * (3 * DD) + 2 * DD + c]);
}

// ===========================================================================
// Flash-style causal attention: one wave per (b, h, 16-query tile).
// ===========================================================================
__global__ void attn_kernel(const __bf16* __restrict__ qkh,
                            const __bf16* __restrict__ vt,
                            __bf16* __restrict__ o) {
    __shared__ __align__(16) __bf16 plds[8][16][32];

    int wib    = threadIdx.x >> 5;
    int lane   = threadIdx.x & 31;
    int lane16 = lane & 15;
    int hi     = lane >> 4;

    int wid = blockIdx.x * (blockDim.x >> 5) + wib;
    const int QT = SS / 16;
    if (wid >= BB * HH * QT) return;
    int qt = wid % QT;
    int h  = (wid / QT) % HH;
    int b  = wid / (QT * HH);
    int q0 = qt << 4;

    const size_t LD2 = 2 * DD;
    const float scale = 0.17677669529663689f;  // 1/sqrt(32)

    v16bf qa;
    {
        const __bf16* qp = qkh + (size_t)(b * SS + q0 + lane16) * LD2 + h * HDD;
        qa = cat8(*(const v8bf*)(qp + (hi << 4)),
                  *(const v8bf*)(qp + 8 + (hi << 4)));
    }

    float mrow[8], lrow[8];
#pragma unroll
    for (int v = 0; v < 8; ++v) { mrow[v] = -1e30f; lrow[v] = 0.0f; }
    v8f oc0 = zero8(), oc1 = zero8();

    for (int k0 = 0; k0 < q0 + 16; k0 += 32) {
        v16bf kf0 = *(const v16bf*)(qkh + (size_t)(b * SS + k0 + lane16) * LD2
                                    + DD + h * HDD + (hi << 4));
        v16bf kf1 = *(const v16bf*)(qkh + (size_t)(b * SS + k0 + 16 + lane16) * LD2
                                    + DD + h * HDD + (hi << 4));
        v8f s0 = __builtin_amdgcn_wmma_f32_16x16x32_bf16(false, qa, false, kf0, (short)0, zero8(), false, false);
        v8f s1 = __builtin_amdgcn_wmma_f32_16x16x32_bf16(false, qa, false, kf1, (short)0, zero8(), false, false);

#pragma unroll
        for (int v = 0; v < 8; ++v) {
            int qrow = q0 + (hi << 3) + v;
            float x0 = (k0 + lane16      <= qrow) ? s0[v] * scale : -1e30f;
            float x1 = (k0 + 16 + lane16 <= qrow) ? s1[v] * scale : -1e30f;
            float t = fmaxf(x0, x1);
#pragma unroll
            for (int m = 1; m < 16; m <<= 1) t = fmaxf(t, __shfl_xor(t, m, 32));
            float mn = fmaxf(mrow[v], t);
            float al = __expf(mrow[v] - mn);
            mrow[v] = mn;
            float p0 = __expf(x0 - mn);
            float p1 = __expf(x1 - mn);
            float rs = p0 + p1;
#pragma unroll
            for (int m = 1; m < 16; m <<= 1) rs += __shfl_xor(rs, m, 32);
            lrow[v] = lrow[v] * al + rs;
            oc0[v] *= al;
            oc1[v] *= al;
            int r = (hi << 3) + v;
            plds[wib][r][lane16]      = f2bf(p0);
            plds[wib][r][lane16 + 16] = f2bf(p1);
        }
        asm volatile("s_wait_dscnt 0" ::: "memory");   // LDS transpose handoff

        v16bf pa = cat8(*(const v8bf*)&plds[wib][lane16][(hi << 4)],
                        *(const v8bf*)&plds[wib][lane16][8 + (hi << 4)]);

        v16bf vf0 = *(const v16bf*)(vt + ((size_t)((b * HH + h) * HDD + lane16)) * SS
                                    + k0 + (hi << 4));
        v16bf vf1 = *(const v16bf*)(vt + ((size_t)((b * HH + h) * HDD + 16 + lane16)) * SS
                                    + k0 + (hi << 4));
        oc0 = __builtin_amdgcn_wmma_f32_16x16x32_bf16(false, pa, false, vf0, (short)0, oc0, false, false);
        oc1 = __builtin_amdgcn_wmma_f32_16x16x32_bf16(false, pa, false, vf1, (short)0, oc1, false, false);
    }

#pragma unroll
    for (int v = 0; v < 8; ++v) {
        int r = (hi << 3) + v;
        float inv = 1.0f / lrow[v];
        size_t base = (size_t)(b * SS + q0 + r) * DD + h * HDD;
        o[base + lane16]      = f2bf(oc0[v] * inv);
        o[base + 16 + lane16] = f2bf(oc1[v] * inv);
    }
}

// ===========================================================================
// TDM warm-up: DMA 32-row slabs of the bf16 head weight (Wt layout [V,K=512])
// into LDS via the Tensor Data Mover -> pulls slabs through L2 ahead of the
// dominant head GEMM. D# built per CDNA5 ISA sec. 8 (2D tensor, 2B elems).
// ===========================================================================
__global__ void tdm_warm_kernel(const __bf16* __restrict__ hdt) {
#if __has_builtin(__builtin_amdgcn_tensor_load_to_lds)
    __shared__ __align__(128) __bf16 stage[32 * 512];
    unsigned long long ga = (unsigned long long)(size_t)hdt
                          + (unsigned long long)blockIdx.x * (32ull * 512ull * 2ull);
    unsigned lds_addr = (unsigned)(size_t)&stage[0];

    v4u g0;
    g0.x = 1u;                                        // count=1 (valid user D#)
    g0.y = lds_addr;                                  // lds_addr [63:32]
    g0.z = (unsigned)(ga & 0xffffffffull);            // global_addr [95:64]
    g0.w = (unsigned)((ga >> 32) & 0x01ffffffull)     // global_addr [120:96]
         | (2u << 30);                                // type=2 ("image")
    v8i g1;
    g1[0] = 0x00010000;                               // data_size=1 (2 bytes)
    g1[1] = (int)((512u & 0xffffu) << 16);            // tensor_dim0 lo16 (=512)
    g1[2] = (int)((32000u & 0xffffu) << 16);          // dim0 hi16=0 | dim1 lo16
    g1[3] = (int)(512u << 16);                        // dim1 hi16=0 | tile_dim0=512
    g1[4] = 32;                                       // tile_dim1=32 | tile_dim2=0
    g1[5] = 512;                                      // tensor_dim0_stride lo32
    g1[6] = 0;                                        // stride hi16 | dim1_stride
    g1[7] = 0;
    v4i g2; g2[0] = 0; g2[1] = 0; g2[2] = 0; g2[3] = 0;
    v4i g3; g3[0] = 0; g3[1] = 0; g3[2] = 0; g3[3] = 0;
#if __has_include(<hip/amd_detail/amd_gfx1250_TDM.h>)
    v8i g4;
#pragma unroll
    for (int i = 0; i < 8; ++i) g4[i] = 0;
    __builtin_amdgcn_tensor_load_to_lds(g0, g1, g2, g3, g4, 0);
#else
    __builtin_amdgcn_tensor_load_to_lds(g0, g1, g2, g3, 0);
#endif
    __builtin_amdgcn_s_wait_tensorcnt(0);
#else
    // fallback: plain cacheline prefetch of the slab
    const char* pp = (const char*)hdt + (size_t)blockIdx.x * 32768 + threadIdx.x * 1024;
    __builtin_prefetch(pp, 0, 0);
#endif
}

// ===========================================================================
// Host-side orchestration
// ===========================================================================
static inline int gemm_blocks(int M, int N) { return ((M / 32) * (N / 64) + 7) / 8; }

extern "C" void kernel_launch(void* const* d_in, const int* in_sizes, int n_in,
                              void* d_out, int out_size, void* d_ws, size_t ws_size,
                              hipStream_t stream) {
    const int*   tokens = (const int*)  d_in[0];
    const float* emb    = (const float*)d_in[1];
    const float* qkv_w  = (const float*)d_in[2];
    const float* qkv_b  = (const float*)d_in[3];
    const float* ow     = (const float*)d_in[4];
    const float* ob     = (const float*)d_in[5];
    const float* ln1_g  = (const float*)d_in[6];
    const float* ln1_b  = (const float*)d_in[7];
    const float* ln2_g  = (const float*)d_in[8];
    const float* ln2_b  = (const float*)d_in[9];
    const float* f1_w   = (const float*)d_in[10];
    const float* f1_b   = (const float*)d_in[11];
    const float* f2_w   = (const float*)d_in[12];
    const float* f2_b   = (const float*)d_in[13];
    const float* head_w = (const float*)d_in[14];
    const float* head_b = (const float*)d_in[15];
    float* out = (float*)d_out;

    char* p = (char*)d_ws;
    float*  x    = (float*) p; p += (size_t)BS * DD * 4;
    float*  qkv  = (float*) p; p += (size_t)BS * 3 * DD * 4;
    __bf16* hb   = (__bf16*)p; p += (size_t)BS * DD * 2;
    __bf16* qkh  = (__bf16*)p; p += (size_t)BS * 2 * DD * 2;
    __bf16* vt   = (__bf16*)p; p += (size_t)BS * DD * 2;
    __bf16* ob16 = (__bf16*)p; p += (size_t)BS * DD * 2;
    __bf16* m1   = (__bf16*)p; p += (size_t)BS * DFF * 2;
    __bf16* xh   = (__bf16*)p; p += (size_t)BS * DD * 2;
    __bf16* qkvwt= (__bf16*)p; p += (size_t)LL * 3 * DD * DD * 2;
    __bf16* owt  = (__bf16*)p; p += (size_t)LL * DD * DD * 2;
    __bf16* f1t  = (__bf16*)p; p += (size_t)LL * DFF * DD * 2;
    __bf16* f2t  = (__bf16*)p; p += (size_t)LL * DD * DFF * 2;
    __bf16* hdt  = (__bf16*)p; p += (size_t)VV * DD * 2;

    // --- one-time weight transpose+convert ---
    for (int l = 0; l < LL; ++l) {
        wt_kernel<<<dim3(3 * DD / 32, DD / 32), 256, 0, stream>>>(
            qkv_w + (size_t)l * DD * 3 * DD, qkvwt + (size_t)l * 3 * DD * DD, DD, 3 * DD);
        wt_kernel<<<dim3(DD / 32, DD / 32), 256, 0, stream>>>(
            ow + (size_t)l * DD * DD, owt + (size_t)l * DD * DD, DD, DD);
        wt_kernel<<<dim3(DFF / 32, DD / 32), 256, 0, stream>>>(
            f1_w + (size_t)l * DD * DFF, f1t + (size_t)l * DFF * DD, DD, DFF);
        wt_kernel<<<dim3(DD / 32, DFF / 32), 256, 0, stream>>>(
            f2_w + (size_t)l * DFF * DD, f2t + (size_t)l * DD * DFF, DFF, DD);
    }
    wt_kernel<<<dim3(VV / 32, DD / 32), 256, 0, stream>>>(head_w, hdt, DD, VV);

    embed_kernel<<<BS, 128, 0, stream>>>(tokens, emb, x);

    const int ropeN = BS * HH * (HDD / 2);
    for (int l = 0; l < LL; ++l) {
        ln_kernel<<<BS / 8, 256, 0, stream>>>(x, ln1_g + l * DD, ln1_b + l * DD, hb);

        gemm_kernel<DD, false, false, false><<<gemm_blocks(BS, 3 * DD), 256, 0, stream>>>(
            hb, qkvwt + (size_t)l * 3 * DD * DD, qkv_b + (size_t)l * 3 * DD,
            nullptr, qkv, BS, 3 * DD);

        ropeconv_kernel<<<(ropeN + 255) / 256, 256, 0, stream>>>(qkv, qkh);
        vtrans_kernel<<<(BS * DD + 255) / 256, 256, 0, stream>>>(qkv, vt);

        attn_kernel<<<(BB * HH * (SS / 16)) / 8, 256, 0, stream>>>(qkh, vt, ob16);

        gemm_kernel<DD, false, true, false><<<gemm_blocks(BS, DD), 256, 0, stream>>>(
            ob16, owt + (size_t)l * DD * DD, ob + (size_t)l * DD,
            x, x, BS, DD);

        ln_kernel<<<BS / 8, 256, 0, stream>>>(x, ln2_g + l * DD, ln2_b + l * DD, hb);

        gemm_kernel<DD, true, false, true><<<gemm_blocks(BS, DFF), 256, 0, stream>>>(
            hb, f1t + (size_t)l * DFF * DD, f1_b + (size_t)l * DFF,
            nullptr, m1, BS, DFF);

        gemm_kernel<DFF, false, true, false><<<gemm_blocks(BS, DD), 256, 0, stream>>>(
            m1, f2t + (size_t)l * DD * DFF, f2_b + (size_t)l * DD,
            x, x, BS, DD);
    }

    cvt_kernel<<<(BS * DD + 255) / 256, 256, 0, stream>>>(x, xh, BS * DD);

    // TDM warm of head weights into L2, then the dominant head GEMM.
    tdm_warm_kernel<<<VV / 32, 32, 0, stream>>>(hdt);
    gemm_kernel<DD, false, false, false><<<gemm_blocks(BS, VV), 256, 0, stream>>>(
        xh, hdt, head_b, nullptr, out, BS, VV);
}
